// AttentionBlock_1924145348864
// MI455X (gfx1250) — matrix operand order repaired and checked by reading, not compiled, not run on hardware
//
#include <hip/hip_runtime.h>
#include <hip/hip_bf16.h>

#define T_DIM 4096
#define C_DIM 512
#define N_BATCH 8
#define TC ((size_t)T_DIM * C_DIM)

typedef __attribute__((ext_vector_type(16))) _Float16 v16h;
typedef __attribute__((ext_vector_type(8)))  _Float16 v8h;
typedef __attribute__((ext_vector_type(8)))  float    v8f;

__device__ __forceinline__ v8f wmma32(v16h a, v16h b, v8f c) {
  // D = A(16x32 f16) x B(32x16 f16) + C(16x16 f32)
  return __builtin_amdgcn_wmma_f32_16x16x32_f16(false, a, false, b, (short)0, c,
                                                false, false);
}

__device__ __forceinline__ v8f zero_v8f() {
  v8f z;
#pragma unroll
  for (int i = 0; i < 8; ++i) z[i] = 0.0f;
  return z;
}

// Load one 16x32 f16 operand fragment (A-layout / N x K B-layout):
// lane L -> row (L&15), K-halves at koff=(L>>4)*8 : halves [koff..koff+7] and
// [koff+16..koff+23] -> two 16B loads. p0 points at (row0, k0), ld in halves.
__device__ __forceinline__ v16h ldfrag(const _Float16* p0, int ld) {
  const int lane = (int)(threadIdx.x & 31u);
  const _Float16* p = p0 + (size_t)(lane & 15) * ld + ((lane >> 4) << 3);
  union { v16h v; v8h h[2]; } u;
  u.h[0] = *(const v8h*)(p);
  u.h[1] = *(const v8h*)(p + 16);
  return u.v;
}

// ---------------------------------------------------------------- cvt f32->f16
__global__ __launch_bounds__(256) void cvt_f16(const float* __restrict__ s,
                                               _Float16* __restrict__ d, int n) {
  int i = blockIdx.x * 256 + threadIdx.x;
  if (i < n) d[i] = (_Float16)s[i];
}

// ---------------------------------------------------------------- groupnorm
__global__ __launch_bounds__(256) void gn_stats(const float* __restrict__ x,
                                                float* __restrict__ stats) {
  __shared__ float ssum[256];
  __shared__ float ssq[256];
  const int bid = blockIdx.x;              // n*32 + g, group = 16 ch * 4096
  const float* p = x + (size_t)bid * 16 * T_DIM;
  const int tid = threadIdx.x;
  float s = 0.f, q = 0.f;
  for (int i = tid; i < 16 * T_DIM; i += 256) {
    float v = p[i];
    s += v; q += v * v;
  }
  ssum[tid] = s; ssq[tid] = q;
  __syncthreads();
  for (int st = 128; st > 0; st >>= 1) {
    if (tid < st) { ssum[tid] += ssum[tid + st]; ssq[tid] += ssq[tid + st]; }
    __syncthreads();
  }
  if (tid == 0) {
    const float inv = 1.0f / (16.0f * T_DIM);
    float mean = ssum[0] * inv;
    float var = ssq[0] * inv - mean * mean;
    stats[bid * 2 + 0] = mean;
    stats[bid * 2 + 1] = rsqrtf(var + 1e-5f);
  }
}

// normalize + transpose: hT[n][t][c] (f16, c contiguous)
__global__ __launch_bounds__(256) void gn_apply(const float* __restrict__ x,
                                                const float* __restrict__ w,
                                                const float* __restrict__ b,
                                                const float* __restrict__ stats,
                                                _Float16* __restrict__ hT) {
  __shared__ float tile[32][65];
  const int tid = threadIdx.x;
  const int tB = blockIdx.x * 64, cB = blockIdx.y * 32, nb = blockIdx.z;
  const int tl = tid & 63, rg = tid >> 6;
#pragma unroll
  for (int i = 0; i < 8; ++i) {
    int c = rg * 8 + i;
    int cg = cB + c;
    const float* st = stats + ((size_t)nb * 32 + (cg >> 4)) * 2;
    float mean = st[0], rstd = st[1];
    float v = x[((size_t)nb * C_DIM + cg) * T_DIM + tB + tl];
    tile[c][tl] = (v - mean) * rstd * w[cg] + b[cg];
  }
  __syncthreads();
  const int t = tid >> 2, cg8 = (tid & 3) * 8;
  union { _Float16 h[8]; float4 f4; } u;
#pragma unroll
  for (int e = 0; e < 8; ++e) u.h[e] = (_Float16)tile[cg8 + e][t];
  *(float4*)&hT[((size_t)nb * T_DIM + tB + t) * C_DIM + cB + cg8] = u.f4;
}

// ---------------------------------------------------------------- generic GEMM
// C[n][m] (f16) = scale * sum_k A[m][k]*B[n][k] + bias  (bias on m or n)
// grid = (M/128, N/128, batch), block = 256 (8 waves, 2m x 4n, 64x32 per wave)
template <int BMODE>
__global__ __launch_bounds__(256) void gemm_tn(
    const _Float16* __restrict__ A, int lda, long sA,
    const _Float16* __restrict__ B, int ldb, long sB,
    _Float16* __restrict__ C, int ldc, long sC,
    const float* __restrict__ bias, float scale, int K) {
  __shared__ _Float16 As[128 * 40];
  __shared__ _Float16 Bs[128 * 40];
  const int tid = threadIdx.x, lane = tid & 31, wave = tid >> 5;
  const int mB = blockIdx.x * 128, nB = blockIdx.y * 128;
  A += (size_t)blockIdx.z * sA;
  B += (size_t)blockIdx.z * sB;
  C += (size_t)blockIdx.z * sC;
  const int wm = wave >> 2, wn = wave & 3;   // 2 x 4 waves
  v8f acc[4][2];
#pragma unroll
  for (int mi = 0; mi < 4; ++mi)
#pragma unroll
    for (int ni = 0; ni < 2; ++ni) acc[mi][ni] = zero_v8f();

  const int row = tid >> 1, hh = tid & 1;
  for (int kk = 0; kk < K; kk += 32) {
    __syncthreads();
    *(float4*)&As[row * 40 + hh * 16] =
        *(const float4*)&A[(size_t)(mB + row) * lda + kk + hh * 16];
    *(float4*)&As[row * 40 + hh * 16 + 8] =
        *(const float4*)&A[(size_t)(mB + row) * lda + kk + hh * 16 + 8];
    *(float4*)&Bs[row * 40 + hh * 16] =
        *(const float4*)&B[(size_t)(nB + row) * ldb + kk + hh * 16];
    *(float4*)&Bs[row * 40 + hh * 16 + 8] =
        *(const float4*)&B[(size_t)(nB + row) * ldb + kk + hh * 16 + 8];
    __syncthreads();
    v16h bf[2];
#pragma unroll
    for (int ni = 0; ni < 2; ++ni)
      bf[ni] = ldfrag(&Bs[(wn * 32 + ni * 16) * 40], 40);
#pragma unroll
    for (int mi = 0; mi < 4; ++mi) {
      v16h a = ldfrag(&As[(wm * 64 + mi * 16) * 40], 40);
#pragma unroll
      for (int ni = 0; ni < 2; ++ni) acc[mi][ni] = wmma32(a, bf[ni], acc[mi][ni]);
    }
  }
#pragma unroll
  for (int mi = 0; mi < 4; ++mi)
#pragma unroll
    for (int ni = 0; ni < 2; ++ni) {
      const int nn = nB + wn * 32 + ni * 16 + (lane & 15);
      const int m8 = mB + wm * 64 + mi * 16 + ((lane >> 4) << 3);
      float bv = (BMODE == 2) ? bias[nn] : 0.0f;
      union { _Float16 h[8]; float4 f4; } u;
#pragma unroll
      for (int r = 0; r < 8; ++r) {
        float v = acc[mi][ni][r] * scale;
        if (BMODE == 1) v += bias[m8 + r];
        if (BMODE == 2) v += bv;
        u.h[r] = (_Float16)v;
      }
      *(float4*)&C[(size_t)nn * ldc + m8] = u.f4;
    }
}

// ---------------------------------------------------------------- flash attn
// q/k: [n][t][c] f16 (q pre-scaled by C^-0.5), v: [n][c][t] f16 -> o: [n][t][c]
// block = 128 (4 waves); 16 q rows per block; wave w owns output c-slice
// [w*128, w*128+128). S recomputed per wave; O lives in registers (8 x v8f).
__global__ __launch_bounds__(128) void fa_kernel(const _Float16* __restrict__ qT,
                                                 const _Float16* __restrict__ kT,
                                                 const _Float16* __restrict__ vv,
                                                 _Float16* __restrict__ o) {
  __shared__ _Float16 Qsub[16 * 72];       // 16 q rows x 64 k-cols (+pad)
  __shared__ _Float16 Ksub[64 * 72];       // 64 kv rows x 64 k-cols
  __shared__ _Float16 Ps[4][16 * 72];      // per-wave probabilities (A layout src)
  __shared__ float    Ost[4][16 * 132];    // per-wave output staging

  const int tid = threadIdx.x, lane = tid & 31, wave = tid >> 5;
  const int qBase = blockIdx.x * 16;
  const int nb = blockIdx.y;
  const _Float16* qg = qT + (size_t)nb * TC;
  const _Float16* kg = kT + (size_t)nb * TC;
  const _Float16* vg = vv + (size_t)nb * TC;
  const int ch0 = wave * 128;

  v8f Oa[8];
#pragma unroll
  for (int i = 0; i < 8; ++i) Oa[i] = zero_v8f();
  float mrow[8], lrow[8];
#pragma unroll
  for (int r = 0; r < 8; ++r) { mrow[r] = -1e30f; lrow[r] = 0.0f; }

  for (int kv = 0; kv < T_DIM; kv += 64) {
    v8f S[4];
#pragma unroll
    for (int f = 0; f < 4; ++f) S[f] = zero_v8f();

    for (int kc = 0; kc < C_DIM; kc += 64) {
      __syncthreads();
      for (int idx = tid; idx < 320; idx += 128) {
        if (idx < 64) {
          int rr = idx >> 2, ch = idx & 3;
          *(float4*)&Qsub[rr * 72 + ch * 16] =
              *(const float4*)&qg[(size_t)(qBase + rr) * C_DIM + kc + ch * 16];
          *(float4*)&Qsub[rr * 72 + ch * 16 + 8] =
              *(const float4*)&qg[(size_t)(qBase + rr) * C_DIM + kc + ch * 16 + 8];
        } else {
          int t = idx - 64;
          int rr = t >> 2, ch = t & 3;
          *(float4*)&Ksub[rr * 72 + ch * 16] =
              *(const float4*)&kg[(size_t)(kv + rr) * C_DIM + kc + ch * 16];
          *(float4*)&Ksub[rr * 72 + ch * 16 + 8] =
              *(const float4*)&kg[(size_t)(kv + rr) * C_DIM + kc + ch * 16 + 8];
        }
      }
      __syncthreads();
#pragma unroll
      for (int ks = 0; ks < 64; ks += 32) {
        v16h a = ldfrag(&Qsub[ks], 72);
#pragma unroll
        for (int f = 0; f < 4; ++f) {
          v16h bb = ldfrag(&Ksub[f * 16 * 72 + ks], 72);
          S[f] = wmma32(a, bb, S[f]);
        }
      }
    }

    // online softmax over this 64-wide kv chunk (rows r + 8*(lane>>4))
    float rmax[8];
#pragma unroll
    for (int r = 0; r < 8; ++r) {
      float m = S[0][r];
#pragma unroll
      for (int f = 1; f < 4; ++f) m = fmaxf(m, S[f][r]);
      rmax[r] = m;
    }
#pragma unroll
    for (int msk = 1; msk < 16; msk <<= 1)
#pragma unroll
      for (int r = 0; r < 8; ++r)
        rmax[r] = fmaxf(rmax[r], __shfl_xor(rmax[r], msk, 32));

    float alpha[8], psum[8];
#pragma unroll
    for (int r = 0; r < 8; ++r) {
      float mn = fmaxf(mrow[r], rmax[r]);
      alpha[r] = __expf(mrow[r] - mn);
      mrow[r] = mn;
      psum[r] = 0.0f;
    }
    const int prow = 8 * (lane >> 4);
    const int pcol = lane & 15;
#pragma unroll
    for (int f = 0; f < 4; ++f)
#pragma unroll
      for (int r = 0; r < 8; ++r) {
        float p = __expf(S[f][r] - mrow[r]);
        psum[r] += p;
        Ps[wave][(r + prow) * 72 + f * 16 + pcol] = (_Float16)p;
      }
#pragma unroll
    for (int msk = 1; msk < 16; msk <<= 1)
#pragma unroll
      for (int r = 0; r < 8; ++r) psum[r] += __shfl_xor(psum[r], msk, 32);
#pragma unroll
    for (int r = 0; r < 8; ++r) lrow[r] = lrow[r] * alpha[r] + psum[r];

    // rescale running O, then accumulate P x V (V frags straight from L2)
#pragma unroll
    for (int i = 0; i < 8; ++i)
#pragma unroll
      for (int r = 0; r < 8; ++r) Oa[i][r] *= alpha[r];

    v16h a0 = ldfrag(&Ps[wave][0], 72);    // kv 0..31 (same-wave DS in-order)
    v16h a1 = ldfrag(&Ps[wave][32], 72);   // kv 32..63
#pragma unroll
    for (int ct = 0; ct < 8; ++ct) {
      const _Float16* vp = vg + (size_t)(ch0 + ct * 16) * T_DIM + kv;
      v16h b0 = ldfrag(vp, T_DIM);
      v16h b1 = ldfrag(vp + 32, T_DIM);
      Oa[ct] = wmma32(a0, b0, Oa[ct]);
      Oa[ct] = wmma32(a1, b1, Oa[ct]);
    }
  }

  // normalize, stage through LDS for packed coalesced f16 stores
  float linv[8];
#pragma unroll
  for (int r = 0; r < 8; ++r) linv[r] = 1.0f / lrow[r];
  const int prow = 8 * (lane >> 4);
  const int pcol = lane & 15;
#pragma unroll
  for (int ct = 0; ct < 8; ++ct)
#pragma unroll
    for (int r = 0; r < 8; ++r)
      Ost[wave][(r + prow) * 132 + ct * 16 + pcol] = Oa[ct][r] * linv[r];

  _Float16* og = o + (size_t)nb * TC;
#pragma unroll
  for (int j = 0; j < 8; ++j) {
    int G = j * 32 + lane;
    int rr = G >> 4, cg = G & 15;
    union { _Float16 h[8]; float4 f4; } u;
#pragma unroll
    for (int e = 0; e < 8; ++e)
      u.h[e] = (_Float16)Ost[wave][rr * 132 + cg * 8 + e];
    *(float4*)&og[(size_t)(qBase + rr) * C_DIM + ch0 + cg * 8] = u.f4;
  }
}

// ---------------------------------------------------------------- proj + resid
// out[n][d][t] (f32) = x + sum_c o[t][c]*Wp[d][c] + proj_b[d]
// A = o (m=t), B = Wp (n=d); grid (4096/128, 512/128, 8)
__global__ __launch_bounds__(256) void proj_res(
    const _Float16* __restrict__ A, int lda, long sA,
    const _Float16* __restrict__ B, int ldb,
    float* __restrict__ Out, int ldc, long sC,
    const float* __restrict__ X, const float* __restrict__ bias, int K) {
  __shared__ _Float16 As[128 * 40];
  __shared__ _Float16 Bs[128 * 40];
  const int tid = threadIdx.x, lane = tid & 31, wave = tid >> 5;
  const int mB = blockIdx.x * 128, nB = blockIdx.y * 128;
  A += (size_t)blockIdx.z * sA;
  Out += (size_t)blockIdx.z * sC;
  X += (size_t)blockIdx.z * sC;
  const int wm = wave >> 2, wn = wave & 3;
  v8f acc[4][2];
#pragma unroll
  for (int mi = 0; mi < 4; ++mi)
#pragma unroll
    for (int ni = 0; ni < 2; ++ni) acc[mi][ni] = zero_v8f();

  const int row = tid >> 1, hh = tid & 1;
  for (int kk = 0; kk < K; kk += 32) {
    __syncthreads();
    *(float4*)&As[row * 40 + hh * 16] =
        *(const float4*)&A[(size_t)(mB + row) * lda + kk + hh * 16];
    *(float4*)&As[row * 40 + hh * 16 + 8] =
        *(const float4*)&A[(size_t)(mB + row) * lda + kk + hh * 16 + 8];
    *(float4*)&Bs[row * 40 + hh * 16] =
        *(const float4*)&B[(size_t)(nB + row) * ldb + kk + hh * 16];
    *(float4*)&Bs[row * 40 + hh * 16 + 8] =
        *(const float4*)&B[(size_t)(nB + row) * ldb + kk + hh * 16 + 8];
    __syncthreads();
    v16h bf[2];
#pragma unroll
    for (int ni = 0; ni < 2; ++ni)
      bf[ni] = ldfrag(&Bs[(wn * 32 + ni * 16) * 40], 40);
#pragma unroll
    for (int mi = 0; mi < 4; ++mi) {
      v16h a = ldfrag(&As[(wm * 64 + mi * 16) * 40], 40);
#pragma unroll
      for (int ni = 0; ni < 2; ++ni) acc[mi][ni] = wmma32(a, bf[ni], acc[mi][ni]);
    }
  }
#pragma unroll
  for (int mi = 0; mi < 4; ++mi)
#pragma unroll
    for (int ni = 0; ni < 2; ++ni) {
      const int nn = nB + wn * 32 + ni * 16 + (lane & 15);
      const int m8 = mB + wm * 64 + mi * 16 + ((lane >> 4) << 3);
      const float bv = bias[nn];
      size_t o0 = (size_t)nn * ldc + m8;
      float4 x0 = *(const float4*)&X[o0];
      float4 x1 = *(const float4*)&X[o0 + 4];
      float4 r0, r1;
      r0.x = x0.x + acc[mi][ni][0] + bv;
      r0.y = x0.y + acc[mi][ni][1] + bv;
      r0.z = x0.z + acc[mi][ni][2] + bv;
      r0.w = x0.w + acc[mi][ni][3] + bv;
      r1.x = x1.x + acc[mi][ni][4] + bv;
      r1.y = x1.y + acc[mi][ni][5] + bv;
      r1.z = x1.z + acc[mi][ni][6] + bv;
      r1.w = x1.w + acc[mi][ni][7] + bv;
      *(float4*)&Out[o0] = r0;
      *(float4*)&Out[o0 + 4] = r1;
    }
}

// ---------------------------------------------------------------- launcher
extern "C" void kernel_launch(void* const* d_in, const int* in_sizes, int n_in,
                              void* d_out, int out_size, void* d_ws,
                              size_t ws_size, hipStream_t stream) {
  (void)in_sizes; (void)n_in; (void)out_size; (void)ws_size;
  const float* x      = (const float*)d_in[0];
  const float* gn_w   = (const float*)d_in[1];
  const float* gn_b   = (const float*)d_in[2];
  const float* qkv_w  = (const float*)d_in[3];
  const float* qkv_b  = (const float*)d_in[4];
  const float* proj_w = (const float*)d_in[5];
  const float* proj_b = (const float*)d_in[6];
  float* out = (float*)d_out;

  char* ws = (char*)d_ws;
  size_t off = 0;
  float* stats = (float*)(ws + off); off += (size_t)N_BATCH * 32 * 2 * sizeof(float);
  off = (off + 255) & ~(size_t)255;
  _Float16* wqkv = (_Float16*)(ws + off); off += (size_t)1536 * 512 * 2;
  _Float16* wprj = (_Float16*)(ws + off); off += (size_t)512 * 512 * 2;
  _Float16* hT   = (_Float16*)(ws + off); off += (size_t)N_BATCH * TC * 2;
  _Float16* qTb  = (_Float16*)(ws + off); off += (size_t)N_BATCH * TC * 2;
  _Float16* kTb  = (_Float16*)(ws + off); off += (size_t)N_BATCH * TC * 2;
  _Float16* vb   = (_Float16*)(ws + off); off += (size_t)N_BATCH * TC * 2;
  _Float16* oT   = hT;  // hT dead after the QKV GEMMs -> alias for attention out

  cvt_f16<<<(1536 * 512 + 255) / 256, 256, 0, stream>>>(qkv_w, wqkv, 1536 * 512);
  cvt_f16<<<(512 * 512 + 255) / 256, 256, 0, stream>>>(proj_w, wprj, 512 * 512);
  gn_stats<<<N_BATCH * 32, 256, 0, stream>>>(x, stats);
  gn_apply<<<dim3(T_DIM / 64, C_DIM / 32, N_BATCH), 256, 0, stream>>>(
      x, gn_w, gn_b, stats, hT);

  const float qscale = 0.044194173824159216f;  // 512^-0.5 folded into Q
  // Q: C_out[t][c] = sum_k Wq[c][k] * hT[t][k]   (bias on m=c)
  gemm_tn<1><<<dim3(4, 32, N_BATCH), 256, 0, stream>>>(
      wqkv, 512, 0L, hT, 512, (long)TC, qTb, 512, (long)TC, qkv_b, qscale, 512);
  // K
  gemm_tn<1><<<dim3(4, 32, N_BATCH), 256, 0, stream>>>(
      wqkv + (size_t)512 * 512, 512, 0L, hT, 512, (long)TC, kTb, 512, (long)TC,
      qkv_b + 512, 1.0f, 512);
  // V: swapped operands so v lands as [d][t]; C_out[d][t], bias on n=d
  gemm_tn<2><<<dim3(32, 4, N_BATCH), 256, 0, stream>>>(
      hT, 512, (long)TC, wqkv + (size_t)1024 * 512, 512, 0L, vb, T_DIM,
      (long)TC, qkv_b + 1024, 1.0f, 512);

  fa_kernel<<<dim3(T_DIM / 16, N_BATCH), 128, 0, stream>>>(qTb, kTb, vb, oT);

  proj_res<<<dim3(32, 4, N_BATCH), 256, 0, stream>>>(
      oT, 512, (long)TC, wprj, 512, out, T_DIM, (long)TC, x, proj_b, 512);
}
